// Informer5_3_575525618092
// MI455X (gfx1250) — matrix operand order repaired and checked by reading, compiled, not run on hardware
//
#include <hip/hip_runtime.h>

// ------------------------------------------------------------------
// CDNA5 / gfx1250 Informer forward pass.
// Dominant GEMMs (FFN 1024x2048x4096 / 1024x4096x2048, projections
// 65536x32x32, final 1024x2048x512) use v_wmma_f32_16x16x32_bf16.
// GEMM requirements: M % 128 == 0, K % 32 == 0, N % 8 == 0.
// ------------------------------------------------------------------

typedef __attribute__((ext_vector_type(16))) __bf16 v16bf;
typedef __attribute__((ext_vector_type(8)))  float  v8f;

// ---------------- elementwise conversions ----------------
__global__ void f32_to_bf16_kernel(const float* __restrict__ src,
                                   __bf16* __restrict__ dst, int n) {
  int i = blockIdx.x * blockDim.x + threadIdx.x;
  if (i < n) dst[i] = (__bf16)src[i];
}

// feat (b,c,a) [1024,32,64] -> y (a*1024+b, c) bf16
__global__ void feat_to_ybf_kernel(const float* __restrict__ feat,
                                   __bf16* __restrict__ ybf) {
  int i = blockIdx.x * blockDim.x + threadIdx.x;
  if (i >= 2097152) return;
  int c = i & 31; int row = i >> 5; int b = row & 1023; int a = row >> 10;
  ybf[i] = (__bf16)feat[(size_t)b * 2048 + c * 64 + a];
}

// ---------------- direct conv (small FLOPs) ----------------
__global__ void conv1d_relu_kernel(const float* __restrict__ in,
                                   const float* __restrict__ w,
                                   const float* __restrict__ bias,
                                   float* __restrict__ out,
                                   int Bn, int Cin, int Lin, int Cout, int Lout,
                                   int Kw, int pad, int accMode) {
  int i = blockIdx.x * blockDim.x + threadIdx.x;
  int total = Bn * Cout * Lout;
  if (i >= total) return;
  int o = i % Lout; int r = i / Lout; int co = r % Cout; int b = r / Cout;
  float s = bias[co];
  int p0 = o * 2 - pad;
  int t0 = (p0 < 0) ? -p0 : 0;               // clamp tap range: no per-tap guards
  int t1 = Kw; if (p0 + t1 > Lin) t1 = Lin - p0;
  for (int ci = 0; ci < Cin; ++ci) {
    const float* ip = in + (size_t)(b * Cin + ci) * Lin + p0;
    const float* wp = w + (size_t)(co * Cin + ci) * Kw;
    for (int t = t0; t < t1; ++t) s += ip[t] * wp[t];
  }
  s = fmaxf(s, 0.0f);
  if (accMode) out[i] += s; else out[i] = s;
}

// ---------------- WMMA bf16 GEMM: C = relu?(A@B + bias) ----------------
// Block = 256 threads (8 wave32), 128x64 C tile. Each wave owns 32 rows x
// 32 cols: 2 A fragments x 2 B fragments -> 4 WMMAs per K-step.
__global__ void __launch_bounds__(256)
gemm_bf16_wmma_kernel(const __bf16* __restrict__ A, const __bf16* __restrict__ Bm,
                      const float* __restrict__ bias, float* __restrict__ C,
                      int M, int N, int K, int relu) {
  __shared__ __bf16 As[128][36];  // [m][k]; row stride 18 dwords: conflict-free frags
  __shared__ __bf16 Bs[32][68];   // [k][n]; row stride 34 dwords: conflict-free frags
  const int tid  = threadIdx.x;
  const int lane = tid & 31;
  const int wave = tid >> 5;
  const int m0 = blockIdx.y * 128;
  const int n0 = blockIdx.x * 64;
  const int wm = (wave >> 1) * 32;  // 0,32,64,96
  const int wn = (wave & 1) * 32;   // 0,32
  v8f acc00 = {}, acc01 = {}, acc10 = {}, acc11 = {};
  union Frag { v16bf v; unsigned u[8]; };

  for (int kt = 0; kt < K; kt += 32) {
    // A tile 128x32: 512 chunks of 8 bf16, 2 per thread, branch-free b64 copies
#pragma unroll
    for (int e = tid; e < 512; e += 256) {
      int mm = e >> 2, kk0 = (e & 3) * 8;
      const __bf16* gp = A + (size_t)(m0 + mm) * K + (kt + kk0);
      uint2 q0 = *(const uint2*)gp;
      uint2 q1 = *(const uint2*)(gp + 4);
      *(uint2*)&As[mm][kk0]     = q0;
      *(uint2*)&As[mm][kk0 + 4] = q1;
    }
    // B tile 32x64: 256 chunks of 8 bf16, 1 per thread; N%8==0 so each chunk
    // is entirely in or out of range -> branchless zero-fill select.
    {
      int kk = tid >> 3, nn0 = (tid & 7) * 8;
      uint2 q0 = make_uint2(0u, 0u), q1 = make_uint2(0u, 0u);
      if (n0 + nn0 < N) {
        const __bf16* gp = Bm + (size_t)(kt + kk) * N + (n0 + nn0);
        q0 = *(const uint2*)gp;
        q1 = *(const uint2*)(gp + 4);
      }
      *(uint2*)&Bs[kk][nn0]     = q0;
      *(uint2*)&Bs[kk][nn0 + 4] = q1;
    }
    if (kt + 32 < K) {  // prefetch next K tile (global_prefetch_b8)
      __builtin_prefetch(A + (size_t)(m0 + (tid & 127)) * K + (kt + 32), 0, 0);
      __builtin_prefetch(Bm + (size_t)(kt + 32 + (tid >> 3)) * N + n0, 0, 0);
    }
    __syncthreads();

    // Fragments per ISA 7.12.2 layouts.
    Frag fa0, fa1, fb0, fb1;
    int mrow0 = wm + (lane & 15);
    int mrow1 = mrow0 + 16;
    int khalf = (lane >> 4) * 8;
#pragma unroll
    for (int v = 0; v < 8; ++v) {
      int kb = ((v < 4) ? 0 : 16) + khalf + 2 * (v & 3);   // A: 16x32 bf16
      fa0.u[v] = *(const unsigned*)&As[mrow0][kb];
      fa1.u[v] = *(const unsigned*)&As[mrow1][kb];
      fb0.v[2 * v]     = Bs[kb][wn + (lane & 15)];  // B: lane = N, same K as A
      fb0.v[2 * v + 1] = Bs[kb + 1][wn + (lane & 15)];
      fb1.v[2 * v]     = Bs[kb][wn + 16 + (lane & 15)];
      fb1.v[2 * v + 1] = Bs[kb + 1][wn + 16 + (lane & 15)];
    }
    acc00 = __builtin_amdgcn_wmma_f32_16x16x32_bf16(false, fa0.v, false, fb0.v,
                                                    (short)0, acc00, false, false);
    acc01 = __builtin_amdgcn_wmma_f32_16x16x32_bf16(false, fa0.v, false, fb1.v,
                                                    (short)0, acc01, false, false);
    acc10 = __builtin_amdgcn_wmma_f32_16x16x32_bf16(false, fa1.v, false, fb0.v,
                                                    (short)0, acc10, false, false);
    acc11 = __builtin_amdgcn_wmma_f32_16x16x32_bf16(false, fa1.v, false, fb1.v,
                                                    (short)0, acc11, false, false);
    __syncthreads();
  }

  // D layout: VGPR r -> row +r (lanes 0-15) / +8+r (lanes 16-31), col = lane&15
  int colA = n0 + wn + (lane & 15);
  int colB = colA + 16;
  int rbase = m0 + wm + 8 * (lane >> 4);
  float biasA = (colA < N) ? bias[colA] : 0.0f;
  float biasB = (colB < N) ? bias[colB] : 0.0f;
#pragma unroll
  for (int r = 0; r < 8; ++r) {
    int row0 = rbase + r;
    int row1 = row0 + 16;
    if (colA < N) {
      float v0 = acc00[r] + biasA; if (relu) v0 = fmaxf(v0, 0.0f);
      C[(size_t)row0 * N + colA] = v0;
      float v1 = acc10[r] + biasA; if (relu) v1 = fmaxf(v1, 0.0f);
      C[(size_t)row1 * N + colA] = v1;
    }
    if (colB < N) {
      float v0 = acc01[r] + biasB; if (relu) v0 = fmaxf(v0, 0.0f);
      C[(size_t)row0 * N + colB] = v0;
      float v1 = acc11[r] + biasB; if (relu) v1 = fmaxf(v1, 0.0f);
      C[(size_t)row1 * N + colB] = v1;
    }
  }
}

// ---------------- ProbSparse attention pieces ----------------
// M[ba,h,l] = max_s q.k_samp - sum_s q.k_samp / 1024
__global__ void prob_M_kernel(const float* __restrict__ Pq, const float* __restrict__ Pk,
                              const int* __restrict__ sidx, float* __restrict__ Mout) {
  int i = blockIdx.x * blockDim.x + threadIdx.x;
  if (i >= 524288) return;                 // 64*8*1024
  int l = i & 1023; int rem = i >> 10; int h = rem & 7; int ba = rem >> 3;
  size_t qb = (size_t)ba * 32768 + (size_t)l * 32 + h * 4;
  float q0 = Pq[qb], q1 = Pq[qb + 1], q2 = Pq[qb + 2], q3 = Pq[qb + 3];
  float mx = -3.4e38f, sm = 0.0f;
  const int* sp = sidx + l * 35;
  for (int s = 0; s < 35; ++s) {
    int j = sp[s];
    size_t kb = (size_t)ba * 32768 + (size_t)j * 32 + h * 4;
    float d = q0 * Pk[kb] + q1 * Pk[kb + 1] + q2 * Pk[kb + 2] + q3 * Pk[kb + 3];
    mx = fmaxf(mx, d); sm += d;
  }
  Mout[i] = mx - sm * (1.0f / 1024.0f);
}

// top-35 per (ba,h), lowest index wins ties (lax.top_k). Destroys Mbuf.
__global__ void topk_kernel(float* __restrict__ Mbuf, int* __restrict__ topidx) {
  int t = blockIdx.x * blockDim.x + threadIdx.x;
  if (t >= 512) return;
  float* row = Mbuf + (size_t)t * 1024;
  int* out = topidx + t * 35;
  for (int u = 0; u < 35; ++u) {
    float best = -3.4e38f; int bi = 0;
    for (int l = 0; l < 1024; ++l) {
      float v = row[l];
      if (v > best) { best = v; bi = l; }
    }
    out[u] = bi;
    row[bi] = -3.4e38f;
  }
}

__global__ void vmean_kernel(const float* __restrict__ Pv, float* __restrict__ vmean) {
  int t = blockIdx.x * blockDim.x + threadIdx.x;
  if (t >= 2048) return;                   // (ba*8+h)*4+d
  int d = t & 3; int rem = t >> 2; int h = rem & 7; int ba = rem >> 3;
  size_t base = (size_t)ba * 32768 + h * 4 + d;
  float s = 0.0f;
  for (int l = 0; l < 1024; ++l) s += Pv[base + (size_t)l * 32];
  vmean[t] = s * (1.0f / 1024.0f);
}

__global__ void ctx_fill_kernel(const float* __restrict__ vmean, float* __restrict__ ctx) {
  int i = blockIdx.x * blockDim.x + threadIdx.x;
  if (i >= 2097152) return;
  int c = i & 31; int row = i >> 5; int ba = row >> 10;
  ctx[i] = vmean[ba * 32 + c];
}

// one wave32 per (ba,h,u): softmax(q_red.k / 2) @ v -> scatter into ctx
__global__ void attn_update_kernel(const float* __restrict__ Pq, const float* __restrict__ Pk,
                                   const float* __restrict__ Pv, const int* __restrict__ topidx,
                                   float* __restrict__ ctx) {
  int gt = blockIdx.x * blockDim.x + threadIdx.x;
  int gw = gt >> 5; int lane = gt & 31;
  if (gw >= 64 * 8 * 35) return;
  int u = gw % 35; int rem = gw / 35; int h = rem & 7; int ba = rem >> 3;
  int ti = topidx[(ba * 8 + h) * 35 + u];
  size_t qb = (size_t)ba * 32768 + (size_t)ti * 32 + h * 4;
  float q0 = Pq[qb], q1 = Pq[qb + 1], q2 = Pq[qb + 2], q3 = Pq[qb + 3];
  size_t kbase = (size_t)ba * 32768 + h * 4;
  float mx = -3.4e38f;
  for (int l = lane; l < 1024; l += 32) {
    size_t kb = kbase + (size_t)l * 32;
    float s = 0.5f * (q0 * Pk[kb] + q1 * Pk[kb + 1] + q2 * Pk[kb + 2] + q3 * Pk[kb + 3]);
    mx = fmaxf(mx, s);
  }
  for (int off = 16; off > 0; off >>= 1) mx = fmaxf(mx, __shfl_xor(mx, off, 32));
  float es = 0.f, u0 = 0.f, u1 = 0.f, u2 = 0.f, u3 = 0.f;
  for (int l = lane; l < 1024; l += 32) {
    size_t kb = kbase + (size_t)l * 32;
    float s = 0.5f * (q0 * Pk[kb] + q1 * Pk[kb + 1] + q2 * Pk[kb + 2] + q3 * Pk[kb + 3]);
    float w = __expf(s - mx);
    es += w;
    u0 += w * Pv[kb]; u1 += w * Pv[kb + 1]; u2 += w * Pv[kb + 2]; u3 += w * Pv[kb + 3];
  }
  for (int off = 16; off > 0; off >>= 1) {
    es += __shfl_xor(es, off, 32);
    u0 += __shfl_xor(u0, off, 32);
    u1 += __shfl_xor(u1, off, 32);
    u2 += __shfl_xor(u2, off, 32);
    u3 += __shfl_xor(u3, off, 32);
  }
  if (lane == 0) {
    float inv = 1.0f / es;
    float4 r = make_float4(u0 * inv, u1 * inv, u2 * inv, u3 * inv);
    *(float4*)(ctx + qb) = r;
  }
}

// ---------------- BatchNorm (population var) with residual ----------------
// layout 0: src row-major (a*1024+b, 32); layout 1: src same (b,c,a) as feat.
__global__ void bn_stats_kernel(const float* __restrict__ src, int layout,
                                const float* __restrict__ feat,
                                float* __restrict__ mean, float* __restrict__ var) {
  __shared__ float s1[256], s2[256];
  int c = blockIdx.x;
  float sum = 0.f, sq = 0.f;
  for (int e = threadIdx.x; e < 65536; e += 256) {
    int b = e >> 6, a = e & 63;
    size_t fi = (size_t)b * 2048 + c * 64 + a;
    float t = (layout == 0) ? src[(size_t)(a * 1024 + b) * 32 + c] : src[fi];
    t += feat[fi];
    sum += t; sq += t * t;
  }
  s1[threadIdx.x] = sum; s2[threadIdx.x] = sq;
  __syncthreads();
  for (int o = 128; o > 0; o >>= 1) {
    if (threadIdx.x < o) { s1[threadIdx.x] += s1[threadIdx.x + o]; s2[threadIdx.x] += s2[threadIdx.x + o]; }
    __syncthreads();
  }
  if (threadIdx.x == 0) {
    float mu = s1[0] * (1.0f / 65536.0f);
    mean[c] = mu;
    var[c] = s2[0] * (1.0f / 65536.0f) - mu * mu;
  }
}

__global__ void bn_apply_kernel(const float* __restrict__ src, int layout,
                                float* __restrict__ feat,
                                const float* __restrict__ mean, const float* __restrict__ var,
                                const float* __restrict__ g, const float* __restrict__ beta) {
  int i = blockIdx.x * blockDim.x + threadIdx.x;
  if (i >= 2097152) return;
  int a = i & 63; int rem = i >> 6; int c = rem & 31; int b = rem >> 5;
  float t = (layout == 0) ? src[(size_t)(a * 1024 + b) * 32 + c] : src[i];
  t += feat[i];
  feat[i] = (t - mean[c]) * rsqrtf(var[c] + 1e-5f) * g[c] + beta[c];
}

// ------------------------------------------------------------------
extern "C" void kernel_launch(void* const* d_in, const int* in_sizes, int n_in,
                              void* d_out, int out_size, void* d_ws, size_t ws_size,
                              hipStream_t stream) {
  (void)in_sizes; (void)n_in; (void)out_size; (void)ws_size;
  const float* x = (const float*)d_in[0];
  const int* sidx_all = (const int*)d_in[1];
  auto cp = [&](int i, int j) -> const float* { return (const float*)d_in[2 + i * 6 + j]; };
  auto ap = [&](int j, int t) -> const float* { return (const float*)d_in[50 + j * 16 + t]; };
  const float* FW = (const float*)d_in[98];
  const float* Fb = (const float*)d_in[99];

  constexpr size_t MB = 1024ull * 1024ull;
  char* ws = (char*)d_ws;
  float*  feat   = (float*)(ws + 0);          // 8 MB (1024,32,64)
  float*  Pq     = (float*)(ws + 8 * MB);     // 8 MB (65536,32)  (reused as yo)
  float*  Pk     = (float*)(ws + 16 * MB);    // 8 MB              (reused as f2)
  float*  Pv     = (float*)(ws + 24 * MB);    // 8 MB
  float*  ctx    = (float*)(ws + 32 * MB);    // 8 MB
  float*  Mbuf   = (float*)(ws + 40 * MB);    // 2 MB (64*8*1024)
  int*    topidx = (int*)  (ws + 42 * MB);    // 512*35 ints
  float*  vmean  = (float*)(ws + 43 * MB);    // 2048 floats
  float*  bnmean = (float*)(ws + 43 * MB + 64 * 1024);
  float*  bnvar  = (float*)(ws + 43 * MB + 128 * 1024);
  __bf16* ybf    = (__bf16*)(ws + 44 * MB);   // 4 MB  (2M bf16)  y / ctx_bf / f_bf
  __bf16* hbf    = (__bf16*)(ws + 48 * MB);   // 8 MB  (4M bf16)
  float*  hbuf   = (float*)(ws + 56 * MB);    // 16 MB (1024,4096)
  __bf16* wq_bf  = (__bf16*)(ws + 72 * MB);   // 4 x 1024 bf16
  __bf16* wk_bf  = wq_bf + 1024;
  __bf16* wv_bf  = wq_bf + 2048;
  __bf16* wo_bf  = wq_bf + 3072;
  __bf16* F1wbf  = (__bf16*)(ws + 73 * MB);   // 16 MB (2048x4096)
  __bf16* F2wbf  = (__bf16*)(ws + 89 * MB);   // 16 MB (4096x2048)
  __bf16* Fwbf   = (__bf16*)(ws + 105 * MB);  // 2 MB  (2048x512)
  float*  t1     = (float*)(ws + 107 * MB);   // 4 MB (1024,4,256)
  float*  t2     = (float*)(ws + 111 * MB);   // 8 MB (1024,16,128)

  auto grid1d = [](int n) { return dim3((unsigned)((n + 255) / 256)); };

  // -------- conv blocks (sum of 8) --------
  for (int i = 0; i < 8; ++i) {
    int k0 = 256 >> i;
    conv1d_relu_kernel<<<grid1d(1024 * 4 * 256), 256, 0, stream>>>(
        x, cp(i, 0), cp(i, 1), t1, 1024, 1, 512, 4, 256, k0, (k0 - 2) / 2, 0);
    conv1d_relu_kernel<<<grid1d(1024 * 16 * 128), 256, 0, stream>>>(
        t1, cp(i, 2), cp(i, 3), t2, 1024, 4, 256, 16, 128, 24, 11, 0);
    conv1d_relu_kernel<<<grid1d(1024 * 32 * 64), 256, 0, stream>>>(
        t2, cp(i, 4), cp(i, 5), feat, 1024, 16, 128, 32, 64, 24, 11, (i == 0) ? 0 : 1);
  }

  // -------- 3 ProbAttention + FFN blocks --------
  for (int j = 0; j < 3; ++j) {
    f32_to_bf16_kernel<<<grid1d(1024), 256, 0, stream>>>(ap(j, 0), wq_bf, 1024);
    f32_to_bf16_kernel<<<grid1d(1024), 256, 0, stream>>>(ap(j, 2), wk_bf, 1024);
    f32_to_bf16_kernel<<<grid1d(1024), 256, 0, stream>>>(ap(j, 4), wv_bf, 1024);
    f32_to_bf16_kernel<<<grid1d(1024), 256, 0, stream>>>(ap(j, 6), wo_bf, 1024);
    feat_to_ybf_kernel<<<grid1d(2097152), 256, 0, stream>>>(feat, ybf);

    dim3 gproj(1, 512);  // N=32 (1 tile), M=65536 (512 tiles of 128)
    gemm_bf16_wmma_kernel<<<gproj, 256, 0, stream>>>(ybf, wq_bf, ap(j, 1), Pq, 65536, 32, 32, 0);
    gemm_bf16_wmma_kernel<<<gproj, 256, 0, stream>>>(ybf, wk_bf, ap(j, 3), Pk, 65536, 32, 32, 0);
    gemm_bf16_wmma_kernel<<<gproj, 256, 0, stream>>>(ybf, wv_bf, ap(j, 5), Pv, 65536, 32, 32, 0);

    const int* sj = sidx_all + (size_t)j * 1024 * 35;
    prob_M_kernel<<<grid1d(524288), 256, 0, stream>>>(Pq, Pk, sj, Mbuf);
    topk_kernel<<<2, 256, 0, stream>>>(Mbuf, topidx);
    vmean_kernel<<<8, 256, 0, stream>>>(Pv, vmean);
    ctx_fill_kernel<<<grid1d(2097152), 256, 0, stream>>>(vmean, ctx);
    attn_update_kernel<<<grid1d(64 * 8 * 35 * 32), 256, 0, stream>>>(Pq, Pk, Pv, topidx, ctx);

    f32_to_bf16_kernel<<<grid1d(2097152), 256, 0, stream>>>(ctx, ybf, 2097152);
    gemm_bf16_wmma_kernel<<<gproj, 256, 0, stream>>>(ybf, wo_bf, ap(j, 7), Pq, 65536, 32, 32, 1);

    bn_stats_kernel<<<32, 256, 0, stream>>>(Pq, 0, feat, bnmean, bnvar);
    bn_apply_kernel<<<grid1d(2097152), 256, 0, stream>>>(Pq, 0, feat, bnmean, bnvar, ap(j, 8), ap(j, 9));

    // FFN: f(1024,2048) -> 4096 -> 2048
    f32_to_bf16_kernel<<<grid1d(2097152), 256, 0, stream>>>(feat, ybf, 2097152);
    f32_to_bf16_kernel<<<grid1d(8388608), 256, 0, stream>>>(ap(j, 10), F1wbf, 8388608);
    f32_to_bf16_kernel<<<grid1d(8388608), 256, 0, stream>>>(ap(j, 12), F2wbf, 8388608);
    gemm_bf16_wmma_kernel<<<dim3(64, 8), 256, 0, stream>>>(ybf, F1wbf, ap(j, 11), hbuf, 1024, 4096, 2048, 1);
    f32_to_bf16_kernel<<<grid1d(4194304), 256, 0, stream>>>(hbuf, hbf, 4194304);
    gemm_bf16_wmma_kernel<<<dim3(32, 8), 256, 0, stream>>>(hbf, F2wbf, ap(j, 13), Pk, 1024, 2048, 4096, 1);

    bn_stats_kernel<<<32, 256, 0, stream>>>(Pk, 1, feat, bnmean, bnvar);
    bn_apply_kernel<<<grid1d(2097152), 256, 0, stream>>>(Pk, 1, feat, bnmean, bnvar, ap(j, 14), ap(j, 15));
  }

  // -------- final projection to (1024,1,512) --------
  f32_to_bf16_kernel<<<grid1d(2097152), 256, 0, stream>>>(feat, ybf, 2097152);
  f32_to_bf16_kernel<<<grid1d(1048576), 256, 0, stream>>>(FW, Fwbf, 1048576);
  gemm_bf16_wmma_kernel<<<dim3(8, 8), 256, 0, stream>>>(ybf, Fwbf, Fb, (float*)d_out, 1024, 512, 2048, 0);
}
